// AttentionScoreMask_31507880084055
// MI455X (gfx1250) — matrix operand order repaired and checked by reading, not compiled, not run on hardware
//
#include <hip/hip_runtime.h>
#include <hip/hip_bf16.h>

typedef __attribute__((ext_vector_type(16))) __bf16 v16bf;
typedef __attribute__((ext_vector_type(8)))  __bf16 v8bf;
typedef __attribute__((ext_vector_type(8)))  float  v8f;

#define NB   8
#define NH   12
#define NQ   1024
#define NK   1024
#define CDIM 768
#define HD   64
#define TOPK 512

static __device__ __forceinline__ v16bf cat16(v8bf lo, v8bf hi) {
    union { v16bf v; v8bf h[2]; } u;
    u.h[0] = lo; u.h[1] = hi;
    return u.v;
}

// ---------------------------------------------------------------- f32 -> bf16
__global__ void cvt_f32_bf16_kernel(const float* __restrict__ in,
                                    __bf16* __restrict__ out, int n) {
    int i = blockIdx.x * blockDim.x + threadIdx.x;
    if (i < n) out[i] = (__bf16)in[i];
}

// ---------------------------------------------------- projection GEMM (WMMA)
// Out = X @ W^T, X:(8192,768), W:(768,768). One 16x16 tile per wave.
// Output stored remapped to (B, H, Ntok, hd) bf16 for the attention step.
__global__ void proj_gemm_kernel(const __bf16* __restrict__ X,
                                 const __bf16* __restrict__ W,
                                 __bf16* __restrict__ Out) {
    const int lane = threadIdx.x & 31;
    const int wave = threadIdx.x >> 5;
    const int tile = blockIdx.x * 8 + wave;          // 24576 tiles total
    const int m0 = (tile / (CDIM / 16)) * 16;
    const int n0 = (tile % (CDIM / 16)) * 16;
    const int hi16 = (lane >> 4) & 1;

    // A fragment: 16x32 bf16. lanes 0-15: K 0..15; lanes 16-31: K 16..31.
    const __bf16* arow = X + (size_t)(m0 + (lane & 15)) * CDIM + hi16 * 16;
    // B fragment: 32x16. column n of W^T == row n of W -> contiguous in K.
    const __bf16* brow = W + (size_t)(n0 + (lane & 15)) * CDIM + hi16 * 16;

    v8f acc = {};
    for (int k0 = 0; k0 < CDIM; k0 += 32) {
        v8bf alo = *(const v8bf*)(arow + k0);
        v8bf ahi = *(const v8bf*)(arow + k0 + 8);
        v8bf blo = *(const v8bf*)(brow + k0);
        v8bf bhi = *(const v8bf*)(brow + k0 + 8);
        __builtin_prefetch(arow + k0 + 64, 0, 0);    // global_prefetch_b8
        __builtin_prefetch(brow + k0 + 64, 0, 0);
        acc = __builtin_amdgcn_wmma_f32_16x16x32_bf16(
            false, cat16(alo, ahi), false, cat16(blo, bhi),
            (short)0, acc, false, false);
    }

    const int n = n0 + (lane & 15);
    const int h = n >> 6, d = n & 63;
#pragma unroll
    for (int i = 0; i < 8; ++i) {
        int m = m0 + i + hi16 * 8;                   // C/D layout rows
        int bidx = m >> 10, tok = m & 1023;
        Out[((size_t)(bidx * NH + h) * NQ + tok) * HD + d] = (__bf16)acc[i];
    }
}

// ------------------------------------- fused QK^T + softmax + head-mean scores
// One block per (b, 16-query tile). 8 waves x 8 ki-tiles = 1024 keys.
__global__ void attn_scores_kernel(const __bf16* __restrict__ Qp,
                                   const __bf16* __restrict__ Kp,
                                   float* __restrict__ scores) {
    __shared__ float lds_logits[16 * NK];            // 64 KB
    const int tid  = threadIdx.x;
    const int lane = tid & 31;
    const int wave = tid >> 5;
    const int b  = blockIdx.x >> 6;
    const int q0 = (blockIdx.x & 63) << 4;
    const int hi16 = (lane >> 4) & 1;
    const float scale = 0.125f;                      // hd^-0.5 = 1/8

    float* srow = scores + ((size_t)(b * NQ + q0)) * NK;
    const int r  = tid >> 4;                         // softmax row 0..15
    const int c0 = tid & 15;                         // strided column lane

    for (int h = 0; h < NH; ++h) {
        const __bf16* qbase =
            Qp + ((size_t)(b * NH + h) * NQ + q0 + (lane & 15)) * HD + hi16 * 16;
        const __bf16* kbase = Kp + (size_t)(b * NH + h) * NK * HD;

        v8f acc[8];
#pragma unroll
        for (int t = 0; t < 8; ++t) acc[t] = v8f{};

#pragma unroll
        for (int dblk = 0; dblk < 2; ++dblk) {
            const int d0 = dblk * 32;
            v8bf alo = *(const v8bf*)(qbase + d0);
            v8bf ahi = *(const v8bf*)(qbase + d0 + 8);
            v16bf afrag = cat16(alo, ahi);
#pragma unroll
            for (int t = 0; t < 8; ++t) {
                int ki = (wave * 8 + t) * 16 + (lane & 15);
                const __bf16* kr = kbase + (size_t)ki * HD + d0 + hi16 * 16;
                v8bf blo = *(const v8bf*)(kr);
                v8bf bhi = *(const v8bf*)(kr + 8);
                acc[t] = __builtin_amdgcn_wmma_f32_16x16x32_bf16(
                    false, afrag, false, cat16(blo, bhi),
                    (short)0, acc[t], false, false);
            }
        }

        // scatter scaled logits to LDS (16 x 1024 tile)
#pragma unroll
        for (int t = 0; t < 8; ++t) {
            int coln = (wave * 8 + t) * 16 + (lane & 15);
#pragma unroll
            for (int i = 0; i < 8; ++i) {
                lds_logits[(i + hi16 * 8) * NK + coln] = acc[t][i] * scale;
            }
        }
        __syncthreads();

        // row softmax: 16 threads per row, cross-lane reduce in width-16 groups
        float mx = -3.4e38f;
        for (int j = 0; j < NK / 16; ++j)
            mx = fmaxf(mx, lds_logits[r * NK + c0 + j * 16]);
        for (int o = 8; o >= 1; o >>= 1) mx = fmaxf(mx, __shfl_xor(mx, o, 16));

        float sum = 0.f;
        for (int j = 0; j < NK / 16; ++j) {
            float e = __expf(lds_logits[r * NK + c0 + j * 16] - mx);
            lds_logits[r * NK + c0 + j * 16] = e;
            sum += e;
        }
        for (int o = 8; o >= 1; o >>= 1) sum += __shfl_xor(sum, o, 16);
        float inv = 1.0f / (sum * (float)NH);

        if (h == 0) {
            for (int j = 0; j < NK / 16; ++j)
                srow[r * NK + c0 + j * 16] =
                    lds_logits[r * NK + c0 + j * 16] * inv;
        } else {
            for (int j = 0; j < NK / 16; ++j)
                srow[r * NK + c0 + j * 16] +=
                    lds_logits[r * NK + c0 + j * 16] * inv;
        }
        __syncthreads();
    }
}

// ------------------------------------------- per-row bitonic top-K (K = 512)
__global__ void topk_kernel(const float* __restrict__ scores,
                            float* __restrict__ out_mask,
                            float* __restrict__ out_idx) {
    __shared__ float sv[NK];
    __shared__ int   si[NK];
    const int tid = threadIdx.x;
    const int row = blockIdx.x;
    const int b = row >> 10, q = row & 1023;
    const float* srow = scores + (size_t)row * NK;

    for (int j = tid; j < NK; j += 256) { sv[j] = srow[j]; si[j] = j; }
    __syncthreads();

    // sort by (value desc, index asc) -- matches lax.top_k ordering
    for (int k = 2; k <= NK; k <<= 1) {
        for (int j = k >> 1; j > 0; j >>= 1) {
            for (int i = tid; i < NK; i += 256) {
                int ixj = i ^ j;
                if (ixj > i) {
                    float va = sv[i], vb = sv[ixj];
                    int   ia = si[i], ib = si[ixj];
                    bool outOfOrder = (va < vb) || (va == vb && ia > ib);
                    bool descRegion = ((i & k) == 0);
                    if (descRegion ? outOfOrder : !outOfOrder) {
                        sv[i] = vb; sv[ixj] = va;
                        si[i] = ib; si[ixj] = ia;
                    }
                }
            }
            __syncthreads();
        }
    }

    float* orow = out_idx + (size_t)row * TOPK;
    for (int j = tid; j < TOPK; j += 256) orow[j] = (float)si[j];

    if (q == 0) {   // mask[:,0,:] — one write per address, race-free
        for (int j = tid; j < NK; j += 256)
            out_mask[b * NK + si[j]] = (j < TOPK) ? 1.0f : 0.0f;
    }
}

// ---------------------------------------------------------------------- host
extern "C" void kernel_launch(void* const* d_in, const int* in_sizes, int n_in,
                              void* d_out, int out_size, void* d_ws, size_t ws_size,
                              hipStream_t stream) {
    (void)in_sizes; (void)n_in; (void)out_size; (void)ws_size;
    const float* q  = (const float*)d_in[0];
    const float* k  = (const float*)d_in[1];
    const float* Wq = (const float*)d_in[2];
    const float* Wk = (const float*)d_in[3];

    const size_t QK_E = (size_t)NB * NQ * CDIM;   // 6,291,456
    const size_t W_E  = (size_t)CDIM * CDIM;      //   589,824
    const size_t P_E  = (size_t)NB * NH * NQ * HD;// 6,291,456

    __bf16* qbf = (__bf16*)d_ws;
    __bf16* kbf = qbf + QK_E;
    __bf16* wqb = kbf + QK_E;
    __bf16* wkb = wqb + W_E;
    __bf16* Qp  = wkb + W_E;
    __bf16* Kp  = Qp + P_E;
    float*  scores = (float*)(Kp + P_E);          // ~86 MB total workspace

    const int thr = 256;
    cvt_f32_bf16_kernel<<<(int)((QK_E + thr - 1) / thr), thr, 0, stream>>>(q,  qbf, (int)QK_E);
    cvt_f32_bf16_kernel<<<(int)((QK_E + thr - 1) / thr), thr, 0, stream>>>(k,  kbf, (int)QK_E);
    cvt_f32_bf16_kernel<<<(int)((W_E  + thr - 1) / thr), thr, 0, stream>>>(Wq, wqb, (int)W_E);
    cvt_f32_bf16_kernel<<<(int)((W_E  + thr - 1) / thr), thr, 0, stream>>>(Wk, wkb, (int)W_E);

    const int tiles = (NB * NQ / 16) * (CDIM / 16);   // 24576, 8 waves/block
    proj_gemm_kernel<<<tiles / 8, 256, 0, stream>>>(qbf, wqb, Qp);
    proj_gemm_kernel<<<tiles / 8, 256, 0, stream>>>(kbf, wkb, Kp);

    attn_scores_kernel<<<NB * (NQ / 16), 256, 0, stream>>>(Qp, Kp, scores);

    float* out_mask = (float*)d_out;                  // (B, Nk) = 8192 floats
    float* out_idx  = out_mask + (size_t)NB * NK;     // (B, Nq, 512)
    topk_kernel<<<NB * NQ, 256, 0, stream>>>(scores, out_mask, out_idx);
}
